// SparseConvCausalAttention_9165460209906
// MI455X (gfx1250) — matrix operand order repaired and checked by reading, not compiled, not run on hardware
//
#include <hip/hip_runtime.h>

#define HEADS 8
#define IMG 32
#define IMG_SEQ 1024
#define TEXT_LEN 256
#define SEQP 1280
#define BATCH 4
#define DIM 512
#define N_QKV 1536
#define BH 32
#define NEGF -3.402823466e38f
#define SCALE_Q 0.125f

typedef __bf16 bf16_t;
typedef __attribute__((ext_vector_type(16))) __bf16 v16bf;
typedef __attribute__((ext_vector_type(8)))  __bf16 v8bf;
typedef __attribute__((ext_vector_type(8)))  float  v8f;

union FragU { v16bf v; v8bf h[2]; };

static __device__ __forceinline__ bf16_t f2bf(float f) {
  unsigned u = __float_as_uint(f);
  u = (u + 0x7FFFu + ((u >> 16) & 1u)) >> 16;
  unsigned short s = (unsigned short)u;
  bf16_t r;
  __builtin_memcpy(&r, &s, 2);
  return r;
}
static __device__ __forceinline__ float bf2f(bf16_t h) {
  unsigned short s;
  __builtin_memcpy(&s, &h, 2);
  return __uint_as_float(((unsigned)s) << 16);
}

// A-fragment (16x32 bf16): lane holds row (lane&15); sel=lane>>4 picks K-halves
// {0..7,16..23} (sel=0) or {8..15,24..31} (sel=1).  Two 16B loads.
static __device__ __forceinline__ v16bf load_a(const bf16_t* row, int k0, int sel) {
  FragU f;
  f.h[0] = *(const v8bf*)(row + k0 + sel * 8);
  f.h[1] = *(const v8bf*)(row + k0 + 16 + sel * 8);
  return f.v;
}
// B-fragment (32x16 bf16) from B^T stored row-major [N][K]: lane holds column
// (lane&15); K-halves {0..7,16..23} (sel=0) or {8..15,24..31} (sel=1).  Two 16B loads.
static __device__ __forceinline__ v16bf load_b(const bf16_t* rowT, int k0, int sel) {
  FragU f;
  f.h[0] = *(const v8bf*)(rowT + k0 + sel * 8);
  f.h[1] = *(const v8bf*)(rowT + k0 + 16 + sel * 8);
  return f.v;
}
static __device__ __forceinline__ v8f wmma_bf16(v16bf a, v16bf b, v8f c) {
  return __builtin_amdgcn_wmma_f32_16x16x32_bf16(false, a, false, b, (short)0, c,
                                                 false, false);
}

// ---------------- kernel 0: convert / pad / transpose to bf16 ----------------
__global__ void k_prep(const float* __restrict__ x, const float* __restrict__ Wqkv,
                       const float* __restrict__ Wout, bf16_t* __restrict__ Xp,
                       bf16_t* __restrict__ WqkvT, bf16_t* __restrict__ WoutT) {
  const int NX = BATCH * SEQP * DIM;
  const int NW1 = N_QKV * DIM;
  const int NW2 = DIM * DIM;
  const int total = NX + NW1 + NW2;
  for (int idx = blockIdx.x * blockDim.x + threadIdx.x; idx < total;
       idx += gridDim.x * blockDim.x) {
    if (idx < NX) {
      int r = idx >> 9, c = idx & 511;
      int b = r / SEQP, nn = r - b * SEQP;
      float v = (nn < 1279) ? x[(b * 1279 + nn) * DIM + c] : 0.0f;
      Xp[idx] = f2bf(v);
    } else if (idx < NX + NW1) {
      int t = idx - NX;
      int n = t >> 9, k = t & 511;
      WqkvT[t] = f2bf(Wqkv[k * N_QKV + n]);
    } else {
      int t = idx - NX - NW1;
      int n = t >> 9, k = t & 511;
      WoutT[t] = f2bf(Wout[k * DIM + n]);
    }
  }
}

// -------- kernel 1: QKV GEMM [5120x512]x[512x1536], scatter to heads --------
__global__ __launch_bounds__(128) void k_qkv(const bf16_t* __restrict__ Xp,
                                             const bf16_t* __restrict__ WqkvT,
                                             bf16_t* __restrict__ Q,
                                             bf16_t* __restrict__ Kk,
                                             bf16_t* __restrict__ V,
                                             bf16_t* __restrict__ Vt) {
  const int wave = threadIdx.x >> 5, lane = threadIdx.x & 31;
  const int sel = lane >> 4, ln = lane & 15;
  const int m0 = blockIdx.x * 64 + wave * 16;
  const int n0 = blockIdx.y * 64;
  const bf16_t* arow = Xp + (m0 + ln) * DIM;
  v8f acc[4] = {};
  for (int k0 = 0; k0 < DIM; k0 += 32) {
    v16bf a = load_a(arow, k0, sel);
#pragma unroll
    for (int t = 0; t < 4; ++t) {
      const bf16_t* brow = WqkvT + (n0 + t * 16 + ln) * DIM;
      acc[t] = wmma_bf16(a, load_b(brow, k0, sel), acc[t]);
    }
  }
#pragma unroll
  for (int t = 0; t < 4; ++t) {
#pragma unroll
    for (int v = 0; v < 8; ++v) {
      int m = m0 + v + 8 * sel;
      int n = n0 + t * 16 + ln;
      float val = acc[t][v];
      int b = m / SEQP, nn = m - b * SEQP;
      int which = n >> 9, w5 = n & 511, h = w5 >> 6, d = w5 & 63;
      int bh = b * HEADS + h;
      int qi = (bh * SEQP + nn) * 64 + d;
      if (which == 0) {
        Q[qi] = f2bf(val * SCALE_Q);
      } else if (which == 1) {
        Kk[qi] = f2bf(val);
      } else {
        V[qi] = f2bf(val);
        Vt[(bh * 64 + d) * SEQP + nn] = f2bf(val);
      }
    }
  }
}

// -------------- kernel 2: causal text attention (256 x 256 x 64) ------------
__global__ __launch_bounds__(128) void k_text(const bf16_t* __restrict__ Q,
                                              const bf16_t* __restrict__ Kk,
                                              const bf16_t* __restrict__ Vt,
                                              bf16_t* __restrict__ OutAll) {
  __shared__ alignas(16) float sd[64][256];
  __shared__ alignas(16) bf16_t sattn[64][256];
  const int bh = blockIdx.x, rbase = blockIdx.y * 64;
  const int wave = threadIdx.x >> 5, lane = threadIdx.x & 31;
  const int sel = lane >> 4, ln = lane & 15;
  const int b = bh >> 3, h = bh & 7;

  // ---- phase 1: dots = q k^T with causal mask ----
  const bf16_t* qrow = Q + (bh * SEQP + rbase + wave * 16 + ln) * 64;
  v16bf a0 = load_a(qrow, 0, sel);
  v16bf a1 = load_a(qrow, 32, sel);
  for (int jt = 0; jt < 16; ++jt) {
    const bf16_t* krow = Kk + (bh * SEQP + jt * 16 + ln) * 64;
    v8f acc = {};
    acc = wmma_bf16(a0, load_b(krow, 0, sel), acc);
    acc = wmma_bf16(a1, load_b(krow, 32, sel), acc);
#pragma unroll
    for (int v = 0; v < 8; ++v) {
      int ml = wave * 16 + v + 8 * sel;
      int i = rbase + ml;
      int j = jt * 16 + ln;
      sd[ml][j] = (j > i) ? NEGF : acc[v];
    }
  }
  __syncthreads();

  // ---- phase 2: softmax per row (one thread per row) ----
  if (threadIdx.x < 64) {
    float* row = sd[threadIdx.x];
    float mx = NEGF;
    for (int j = 0; j < 256; ++j) mx = fmaxf(mx, row[j]);
    float s = 0.f;
    for (int j = 0; j < 256; ++j) {
      float e = __expf(row[j] - mx);
      row[j] = e;
      s += e;
    }
    float inv = 1.f / s;
    for (int j = 0; j < 256; ++j) row[j] *= inv;
  }
  __syncthreads();
  for (int idx = threadIdx.x; idx < 64 * 256; idx += 128)
    sattn[idx >> 8][idx & 255] = f2bf(sd[idx >> 8][idx & 255]);
  __syncthreads();

  // ---- phase 3: out = attn @ V ----
  const bf16_t* atn = &sattn[wave * 16 + ln][0];
  v8f acc[4] = {};
  for (int j0 = 0; j0 < 256; j0 += 32) {
    v16bf a = load_a(atn, j0, sel);
#pragma unroll
    for (int t = 0; t < 4; ++t) {
      const bf16_t* vtr = Vt + (bh * 64 + t * 16 + ln) * SEQP;
      acc[t] = wmma_bf16(a, load_b(vtr, j0, sel), acc[t]);
    }
  }
#pragma unroll
  for (int t = 0; t < 4; ++t) {
#pragma unroll
    for (int v = 0; v < 8; ++v) {
      int ml = wave * 16 + v + 8 * sel;
      int nn = rbase + ml;
      int d = t * 16 + ln;
      OutAll[(b * SEQP + nn) * DIM + h * 64 + d] = f2bf(acc[t][v]);
    }
  }
}

// ------- kernel 3: image attention (256 text keys + 25 local keys) ----------
__global__ __launch_bounds__(128) void k_img(const bf16_t* __restrict__ Q,
                                             const bf16_t* __restrict__ Kk,
                                             const bf16_t* __restrict__ V,
                                             const bf16_t* __restrict__ Vt,
                                             const unsigned char* __restrict__ mask,
                                             bf16_t* __restrict__ OutAll) {
  __shared__ alignas(16) float sd[64][288];     // 256 i2t + 25 local (+pad)
  __shared__ alignas(16) bf16_t sattn[64][256];
  __shared__ alignas(16) float sloc[64][64];
  const int bh = blockIdx.x, qbase = blockIdx.y * 64;
  const int wave = threadIdx.x >> 5, lane = threadIdx.x & 31;
  const int sel = lane >> 4, ln = lane & 15;
  const int b = bh >> 3, h = bh & 7;

  // ---- phase 1: dots_i2t = q_img k_text^T (WMMA) with text pad mask ----
  const bf16_t* qrow = Q + (bh * SEQP + TEXT_LEN + qbase + wave * 16 + ln) * 64;
  v16bf a0 = load_a(qrow, 0, sel);
  v16bf a1 = load_a(qrow, 32, sel);
  for (int jt = 0; jt < 16; ++jt) {
    const bf16_t* krow = Kk + (bh * SEQP + jt * 16 + ln) * 64;
    v8f acc = {};
    acc = wmma_bf16(a0, load_b(krow, 0, sel), acc);
    acc = wmma_bf16(a1, load_b(krow, 32, sel), acc);
#pragma unroll
    for (int v = 0; v < 8; ++v) {
      int ml = wave * 16 + v + 8 * sel;
      int j = jt * 16 + ln;
      sd[ml][j] = mask[b * TEXT_LEN + j] ? acc[v] : NEGF;
    }
  }

  // ---- phase 2: local 5x5 neighborhood dots (VALU gather) ----
  for (int t = threadIdx.x; t < 64 * 25; t += 128) {
    int ql = t / 25, j = t - ql * 25;
    int i = qbase + ql;
    int y = i >> 5, xx = i & 31;
    int ki = j / 5, kj = j - ki * 5;
    int ny = y + ki - 2, nx = xx + kj - 2;
    float dv = NEGF;
    if (ny >= 0 && ny < IMG && nx >= 0 && nx < IMG) {
      int kidx = ny * IMG + nx;
      if (kidx <= i) {  // causal: masked when q_idx < k_idx (or OOB)
        const bf16_t* qp = Q + (bh * SEQP + TEXT_LEN + i) * 64;
        const bf16_t* kp = Kk + (bh * SEQP + TEXT_LEN + kidx) * 64;
        float s = 0.f;
        for (int d = 0; d < 64; ++d) s += bf2f(qp[d]) * bf2f(kp[d]);
        dv = s;
      }
    }
    sd[ql][256 + j] = dv;
  }
  __syncthreads();

  // ---- phase 3: joint softmax over 281 logits ----
  if (threadIdx.x < 64) {
    float* row = sd[threadIdx.x];
    float mx = NEGF;
    for (int j = 0; j < 281; ++j) mx = fmaxf(mx, row[j]);
    float s = 0.f;
    for (int j = 0; j < 281; ++j) {
      float e = __expf(row[j] - mx);
      row[j] = e;
      s += e;
    }
    float inv = 1.f / s;
    for (int j = 0; j < 281; ++j) row[j] *= inv;
  }
  __syncthreads();

  // ---- phase 4a: local attn @ v_unfold (VALU gather) + bf16 convert ----
  for (int idx = threadIdx.x; idx < 64 * 256; idx += 128)
    sattn[idx >> 8][idx & 255] = f2bf(sd[idx >> 8][idx & 255]);
  {
    int ql = threadIdx.x >> 1;
    int db = (threadIdx.x & 1) * 32;
    int i = qbase + ql;
    int y = i >> 5, xx = i & 31;
    float accv[32] = {};
    for (int j = 0; j < 25; ++j) {
      int ki = j / 5, kj = j - ki * 5;
      int ny = y + ki - 2, nx = xx + kj - 2;
      if (ny >= 0 && ny < IMG && nx >= 0 && nx < IMG) {
        float a = sd[ql][256 + j];  // zero if causally masked
        const bf16_t* vp = V + (bh * SEQP + TEXT_LEN + ny * IMG + nx) * 64 + db;
#pragma unroll
        for (int dd = 0; dd < 32; ++dd) accv[dd] += a * bf2f(vp[dd]);
      }
    }
#pragma unroll
    for (int dd = 0; dd < 32; ++dd) sloc[ql][db + dd] = accv[dd];
  }
  __syncthreads();

  // ---- phase 4b: attn_i2t @ v_text (WMMA), add local, store ----
  const bf16_t* atn = &sattn[wave * 16 + ln][0];
  v8f acc[4] = {};
  for (int j0 = 0; j0 < 256; j0 += 32) {
    v16bf a = load_a(atn, j0, sel);
#pragma unroll
    for (int t = 0; t < 4; ++t) {
      const bf16_t* vtr = Vt + (bh * 64 + t * 16 + ln) * SEQP;
      acc[t] = wmma_bf16(a, load_b(vtr, j0, sel), acc[t]);
    }
  }
#pragma unroll
  for (int t = 0; t < 4; ++t) {
#pragma unroll
    for (int v = 0; v < 8; ++v) {
      int ml = wave * 16 + v + 8 * sel;
      int nn = TEXT_LEN + qbase + ml;
      int d = t * 16 + ln;
      float val = acc[t][v] + sloc[ml][d];
      OutAll[(b * SEQP + nn) * DIM + h * 64 + d] = f2bf(val);
    }
  }
}

// -------- kernel 4: out projection [5120x512]x[512x512] + bias, fp32 --------
__global__ __launch_bounds__(128) void k_out(const bf16_t* __restrict__ OutAll,
                                             const bf16_t* __restrict__ WoutT,
                                             const float* __restrict__ bout,
                                             float* __restrict__ out) {
  const int wave = threadIdx.x >> 5, lane = threadIdx.x & 31;
  const int sel = lane >> 4, ln = lane & 15;
  const int m0 = blockIdx.x * 64 + wave * 16;
  const int n0 = blockIdx.y * 64;
  const bf16_t* arow = OutAll + (m0 + ln) * DIM;
  v8f acc[4] = {};
  for (int k0 = 0; k0 < DIM; k0 += 32) {
    v16bf a = load_a(arow, k0, sel);
#pragma unroll
    for (int t = 0; t < 4; ++t) {
      const bf16_t* brow = WoutT + (n0 + t * 16 + ln) * DIM;
      acc[t] = wmma_bf16(a, load_b(brow, k0, sel), acc[t]);
    }
  }
#pragma unroll
  for (int t = 0; t < 4; ++t) {
#pragma unroll
    for (int v = 0; v < 8; ++v) {
      int m = m0 + v + 8 * sel;
      int n = n0 + t * 16 + ln;
      int b = m / SEQP, nn = m - b * SEQP;
      if (nn < 1279) out[(b * 1279 + nn) * DIM + n] = acc[t][v] + bout[n];
    }
  }
}

extern "C" void kernel_launch(void* const* d_in, const int* in_sizes, int n_in,
                              void* d_out, int out_size, void* d_ws, size_t ws_size,
                              hipStream_t stream) {
  (void)in_sizes; (void)n_in; (void)out_size; (void)ws_size;
  const float* x = (const float*)d_in[0];
  const unsigned char* mask = (const unsigned char*)d_in[1];
  const float* Wqkv = (const float*)d_in[2];
  const float* Wout = (const float*)d_in[3];
  const float* bout = (const float*)d_in[4];
  float* out = (float*)d_out;

  char* p = (char*)d_ws;
  bf16_t* Xp = (bf16_t*)p;     p += (size_t)BATCH * SEQP * DIM * 2;   // 5 MB
  bf16_t* WqkvT = (bf16_t*)p;  p += (size_t)N_QKV * DIM * 2;          // 1.5 MB
  bf16_t* WoutT = (bf16_t*)p;  p += (size_t)DIM * DIM * 2;            // 0.5 MB
  bf16_t* Q = (bf16_t*)p;      p += (size_t)BH * SEQP * 64 * 2;       // 5 MB
  bf16_t* Kk = (bf16_t*)p;     p += (size_t)BH * SEQP * 64 * 2;
  bf16_t* V = (bf16_t*)p;      p += (size_t)BH * SEQP * 64 * 2;
  bf16_t* Vt = (bf16_t*)p;     p += (size_t)BH * 64 * SEQP * 2;
  bf16_t* OutAll = (bf16_t*)p; p += (size_t)BATCH * SEQP * DIM * 2;   // total 32 MB

  k_prep<<<1024, 256, 0, stream>>>(x, Wqkv, Wout, Xp, WqkvT, WoutT);
  k_qkv<<<dim3(80, 24), 128, 0, stream>>>(Xp, WqkvT, Q, Kk, V, Vt);
  k_text<<<dim3(32, 4), 128, 0, stream>>>(Q, Kk, Vt, OutAll);
  k_img<<<dim3(32, 16), 128, 0, stream>>>(Q, Kk, V, Vt, mask, OutAll);
  k_out<<<dim3(80, 8), 128, 0, stream>>>(OutAll, WoutT, bout, out);
}
